// GRU_5102421148040
// MI455X (gfx1250) — matrix operand order repaired and checked by reading, not compiled, not run on hardware
//
#include <hip/hip_runtime.h>
#include <hip/hip_bf16.h>

// ---------------------------------------------------------------------------
// GRU autoregressive sampler for MI455X (gfx1250, wave32, WMMA).
// - bf16 hi/lo split GEMMs (3x v_wmma_f32_16x16x32_bf16 per 16x16x32 tile,
//   two independent accumulator chains) with f32 accumulation.
// - A-tiles staged with GLOBAL_LOAD_ASYNC_TO_LDS_B128 (ASYNCcnt) into a
//   double-buffered LDS tile so the copy overlaps WMMA work.
// - Weights stay in registers straight from L2 (36 MB of weights << 192 MB L2).
// ---------------------------------------------------------------------------

typedef __bf16 bf16;
typedef __attribute__((ext_vector_type(16))) __bf16 bf16x16;
typedef __attribute__((ext_vector_type(8)))  float  f32x8;
typedef __attribute__((ext_vector_type(4)))  unsigned int u32x4;

union FragU { bf16x16 v; u32x4 q[2]; };

#define B_SZ 256
#define H_SZ 512
#define C_SZ 4096
#define L_SZ 48
#define LDS_K 40   // 32 + 8 pad (16B-aligned rows, staggers banks)
#define TILE_ELEMS (64 * LDS_K)

static __device__ inline f32x8 zero8() {
  f32x8 z = {0.f, 0.f, 0.f, 0.f, 0.f, 0.f, 0.f, 0.f};
  return z;
}

static __device__ inline float sigmoidf(float x) {
  return 1.0f / (1.0f + __expf(-x));
}

// A fragment (16x32 bf16) from an LDS tile [64][LDS_K].
// ISA layout: lane&15 = M row; kb=(lane>>4)*8; K = {kb..kb+7, 16+kb..16+kb+7}.
static __device__ inline bf16x16 load_a(const bf16* As, int mo, int lane) {
  int m  = mo + (lane & 15);
  int kb = (lane >> 4) << 3;          // 0 or 8
  const bf16* p = As + m * LDS_K + kb;
  FragU f;
  f.q[0] = *(const u32x4*)(p);        // 16B ds_load_b128
  f.q[1] = *(const u32x4*)(p + 16);
  return f.v;
}

// B fragment (32x16 bf16) directly from global weights W[N,K] row-major.
// ISA layout: lane&15 = N col; kb=(lane>>4)*8; K = {kb..kb+7, 16+kb..16+kb+7}.
static __device__ inline bf16x16 load_b(const bf16* __restrict__ W, int ldk,
                                        int n0, int k0, int lane) {
  int n  = n0 + (lane & 15);
  int kb = k0 + ((lane >> 4) << 3);   // +0 or +8
  const bf16* p = W + (size_t)n * ldk + kb;
  FragU f;
  f.q[0] = *(const u32x4*)(p);        // global_load_b128
  f.q[1] = *(const u32x4*)(p + 16);
  return f.v;
}

static __device__ inline f32x8 wmma_bf16(bf16x16 a, bf16x16 b, f32x8 c) {
  return __builtin_amdgcn_wmma_f32_16x16x32_bf16(false, a, false, b, (short)0, c,
                                                 false, false);
}

// Split-bf16 MAC on two independent accumulator chains:
//   P += Ahi*Bhi ; Q += Ahi*Blo ; Q += Alo*Bhi   (final value = P + Q)
static __device__ inline void mma3s(f32x8& P, f32x8& Q, bf16x16 ah, bf16x16 al,
                                    bf16x16 bh, bf16x16 bl) {
  P = wmma_bf16(ah, bh, P);
  Q = wmma_bf16(ah, bl, Q);
  Q = wmma_bf16(al, bh, Q);
}

// Issue async global->LDS copies for a 64x32 bf16 tile (hi and lo planes).
// 256 threads x 16B = 4KB per plane = one b128 async op per thread per plane.
static __device__ inline void issue_stage(const bf16* __restrict__ Ghi,
                                          const bf16* __restrict__ Glo,
                                          bf16* As_hi, bf16* As_lo,
                                          int ldk, int row0, int k0, int tid) {
  int r = tid >> 2;                   // 0..63
  int c = (tid & 3) << 3;             // 0,8,16,24
  size_t go = (size_t)(row0 + r) * ldk + k0 + c;
  const bf16* gh = Ghi + go;
  const bf16* gl = Glo + go;
  // Flat shared-aperture address: LDS byte offset lives in addr[31:0].
  unsigned ldh = (unsigned)(size_t)(const void*)(As_hi + r * LDS_K + c);
  unsigned ldl = (unsigned)(size_t)(const void*)(As_lo + r * LDS_K + c);
  asm volatile("global_load_async_to_lds_b128 %0, %1, off"
               :: "v"(ldh), "v"(gh) : "memory");
  asm volatile("global_load_async_to_lds_b128 %0, %1, off"
               :: "v"(ldl), "v"(gl) : "memory");
}

static __device__ inline void wait_async_le(int pending) {
  if (pending) asm volatile("s_wait_asynccnt 0x2" ::: "memory");
  else         asm volatile("s_wait_asynccnt 0x0" ::: "memory");
}

// ---------------------------------------------------------------------------
// convert f32 -> (bf16 hi, bf16 lo) split
// ---------------------------------------------------------------------------
__global__ void cvt_split(const float* __restrict__ x, bf16* __restrict__ hi,
                          bf16* __restrict__ lo, size_t n) {
  size_t stride = (size_t)gridDim.x * blockDim.x;
  for (size_t i = (size_t)blockIdx.x * blockDim.x + threadIdx.x; i < n; i += stride) {
    float v = x[i];
    bf16 h = (bf16)v;
    hi[i] = h;
    lo[i] = (bf16)(v - (float)h);
  }
}

__global__ void zero_h(float* __restrict__ hf, bf16* __restrict__ hhi,
                       bf16* __restrict__ hlo, int n) {
  int i = blockIdx.x * blockDim.x + threadIdx.x;
  if (i < n) {
    hf[i]  = 0.0f;
    hhi[i] = (bf16)0.0f;
    hlo[i] = (bf16)0.0f;
  }
}

// ---------------------------------------------------------------------------
// head: codes = sigmoid(X[256,512] @ Wout^T[512,4096] + b_out)
// Block tile 64(M) x 128(N); 8 waves as 4(M) x 2(N); wave tile 16 x 64.
// ---------------------------------------------------------------------------
__global__ __launch_bounds__(256) void head_kernel(
    const bf16* __restrict__ Xhi, const bf16* __restrict__ Xlo,
    const bf16* __restrict__ Whi, const bf16* __restrict__ Wlo,
    const float* __restrict__ bias,
    float* __restrict__ samples, size_t sampRowStride,
    bf16* __restrict__ Chi, bf16* __restrict__ Clo) {
  const int K = H_SZ;
  __shared__ bf16 As_hi[2][TILE_ELEMS];
  __shared__ bf16 As_lo[2][TILE_ELEMS];

  int tid  = threadIdx.x;
  int lane = tid & 31;
  int wave = tid >> 5;
  int bM   = blockIdx.y * 64;
  int bN   = blockIdx.x * 128;
  int mo   = (wave >> 1) * 16;
  int no   = (wave & 1) * 64;

  f32x8 accP[4], accQ[4];
#pragma unroll
  for (int t = 0; t < 4; ++t) { accP[t] = zero8(); accQ[t] = zero8(); }

  issue_stage(Xhi, Xlo, As_hi[0], As_lo[0], K, bM, 0, tid);
  const int nk = K >> 5;              // 16 k-steps
  for (int i = 0; i < nk; ++i) {
    int k0  = i << 5;
    int cur = i & 1;
    bool more = (i + 1 < nk);
    if (more)
      issue_stage(Xhi, Xlo, As_hi[cur ^ 1], As_lo[cur ^ 1], K, bM, k0 + 32, tid);
    wait_async_le(more ? 1 : 0);      // current tile landed; next pair in flight
    __syncthreads();
    bf16x16 ah = load_a(As_hi[cur], mo, lane);
    bf16x16 al = load_a(As_lo[cur], mo, lane);
#pragma unroll
    for (int t = 0; t < 4; ++t) {
      int n0 = bN + no + t * 16;
      bf16x16 bh = load_b(Whi, K, n0, k0, lane);
      bf16x16 bl = load_b(Wlo, K, n0, k0, lane);
      mma3s(accP[t], accQ[t], ah, al, bh, bl);
    }
    __syncthreads();                  // reads done before this buffer is reused
  }

  // Epilogue: D layout -> VGPR g: lanes<16 M=g, lanes>=16 M=g+8; N=lane&15.
  int nlane = lane & 15, half = lane >> 4;
#pragma unroll
  for (int t = 0; t < 4; ++t) {
    int col = bN + no + t * 16 + nlane;
    float bb = bias[col];
#pragma unroll
    for (int g = 0; g < 8; ++g) {
      int row = bM + mo + half * 8 + g;
      float s = sigmoidf(accP[t][g] + accQ[t][g] + bb);
      samples[(size_t)row * sampRowStride + col] = s;
      bf16 h = (bf16)s;
      Chi[(size_t)row * C_SZ + col] = h;
      Clo[(size_t)row * C_SZ + col] = (bf16)(s - (float)h);
    }
  }
}

// ---------------------------------------------------------------------------
// Fused GRU cell. Block tile 64(M) x 32(N over H) -> 64 blocks for occupancy.
// 8 waves as 4(M) x 2(N); wave tile 16 x 16 with 4 gate accumulators x2 chains.
// ---------------------------------------------------------------------------
static __device__ inline void gemm_phase(
    const bf16* __restrict__ Ahi, const bf16* __restrict__ Alo, int K,
    const bf16* __restrict__ Whi, const bf16* __restrict__ Wlo,
    bf16 (*bufh)[TILE_ELEMS], bf16 (*bufl)[TILE_ELEMS],
    int tid, int lane, int bM, int n0, int mo,
    f32x8& Rp, f32x8& Rq, f32x8& Zp, f32x8& Zq, f32x8& Np, f32x8& Nq) {
  issue_stage(Ahi, Alo, bufh[0], bufl[0], K, bM, 0, tid);
  const int nk = K >> 5;
  for (int i = 0; i < nk; ++i) {
    int k0  = i << 5;
    int cur = i & 1;
    bool more = (i + 1 < nk);
    if (more)
      issue_stage(Ahi, Alo, bufh[cur ^ 1], bufl[cur ^ 1], K, bM, k0 + 32, tid);
    wait_async_le(more ? 1 : 0);
    __syncthreads();
    bf16x16 ah = load_a(bufh[cur], mo, lane);
    bf16x16 al = load_a(bufl[cur], mo, lane);
    {   // gate r : W rows 0*H + j
      bf16x16 bh = load_b(Whi, K, 0 * H_SZ + n0, k0, lane);
      bf16x16 bl = load_b(Wlo, K, 0 * H_SZ + n0, k0, lane);
      mma3s(Rp, Rq, ah, al, bh, bl);
    }
    {   // gate z : W rows 1*H + j
      bf16x16 bh = load_b(Whi, K, 1 * H_SZ + n0, k0, lane);
      bf16x16 bl = load_b(Wlo, K, 1 * H_SZ + n0, k0, lane);
      mma3s(Zp, Zq, ah, al, bh, bl);
    }
    {   // gate n : W rows 2*H + j
      bf16x16 bh = load_b(Whi, K, 2 * H_SZ + n0, k0, lane);
      bf16x16 bl = load_b(Wlo, K, 2 * H_SZ + n0, k0, lane);
      mma3s(Np, Nq, ah, al, bh, bl);
    }
    __syncthreads();
  }
}

__global__ __launch_bounds__(256) void gru_kernel(
    const bf16* __restrict__ Codes_hi, const bf16* __restrict__ Codes_lo,
    const bf16* __restrict__ Wih_hi,  const bf16* __restrict__ Wih_lo,
    const bf16* __restrict__ Hhi_in,  const bf16* __restrict__ Hlo_in,
    const bf16* __restrict__ Whh_hi,  const bf16* __restrict__ Whh_lo,
    const float* __restrict__ b_ih,   const float* __restrict__ b_hh,
    const float* __restrict__ Hf_in,
    float* __restrict__ Hf_out, bf16* __restrict__ Hhi_out, bf16* __restrict__ Hlo_out,
    float* __restrict__ hidd_out, size_t hiddRowStride) {
  __shared__ bf16 As_hi[2][TILE_ELEMS];
  __shared__ bf16 As_lo[2][TILE_ELEMS];

  int tid  = threadIdx.x;
  int lane = tid & 31;
  int wave = tid >> 5;
  int bM   = blockIdx.y * 64;
  int bN   = blockIdx.x * 32;
  int mo   = (wave >> 1) * 16;
  int n0   = bN + (wave & 1) * 16;    // this wave's 16 hidden units

  f32x8 Rp = zero8(), Rq = zero8();
  f32x8 Zp = zero8(), Zq = zero8();
  f32x8 INp = zero8(), INq = zero8(); // input path of n-gate
  f32x8 HNp = zero8(), HNq = zero8(); // hidden path of n-gate

  // Phase 1: input path, K = C = 4096 over codes x w_ih
  gemm_phase(Codes_hi, Codes_lo, C_SZ, Wih_hi, Wih_lo, As_hi, As_lo,
             tid, lane, bM, n0, mo, Rp, Rq, Zp, Zq, INp, INq);
  // Phase 2: hidden path, K = H = 512 over h x w_hh
  gemm_phase(Hhi_in, Hlo_in, H_SZ, Whh_hi, Whh_lo, As_hi, As_lo,
             tid, lane, bM, n0, mo, Rp, Rq, Zp, Zq, HNp, HNq);

  // Gate epilogue (torch GRUCell semantics)
  int nlane = lane & 15, half = lane >> 4;
  int j = n0 + nlane;                  // hidden unit index 0..511
  float brI = b_ih[j],            brH = b_hh[j];
  float bzI = b_ih[H_SZ + j],     bzH = b_hh[H_SZ + j];
  float bnI = b_ih[2 * H_SZ + j], bnH = b_hh[2 * H_SZ + j];
#pragma unroll
  for (int g = 0; g < 8; ++g) {
    int row = bM + mo + half * 8 + g;
    float r  = sigmoidf(Rp[g] + Rq[g] + brI + brH);
    float z  = sigmoidf(Zp[g] + Zq[g] + bzI + bzH);
    float n  = tanhf(INp[g] + INq[g] + bnI + r * (HNp[g] + HNq[g] + bnH));
    float hp = Hf_in[(size_t)row * H_SZ + j];
    float hn = (1.0f - z) * n + z * hp;
    hidd_out[(size_t)row * hiddRowStride + j] = hn;
    Hf_out[(size_t)row * H_SZ + j] = hn;
    bf16 hb = (bf16)hn;
    Hhi_out[(size_t)row * H_SZ + j] = hb;
    Hlo_out[(size_t)row * H_SZ + j] = (bf16)(hn - (float)hb);
  }
}

// ---------------------------------------------------------------------------
// Host-side orchestration
// ---------------------------------------------------------------------------
extern "C" void kernel_launch(void* const* d_in, const int* in_sizes, int n_in,
                              void* d_out, int out_size, void* d_ws, size_t ws_size,
                              hipStream_t stream) {
  const float* noise = (const float*)d_in[0];
  const float* w_ih  = (const float*)d_in[1];
  const float* b_ih  = (const float*)d_in[2];
  const float* w_hh  = (const float*)d_in[3];
  const float* b_hh  = (const float*)d_in[4];
  const float* w_out = (const float*)d_in[5];
  const float* b_out = (const float*)d_in[6];
  (void)in_sizes; (void)n_in; (void)out_size; (void)ws_size;

  float* samples = (float*)d_out;                              // [B, L, C]
  float* hiddens = samples + (size_t)B_SZ * L_SZ * C_SZ;       // [B, L, H]

  char* ws = (char*)d_ws;
  size_t off = 0;
  auto take = [&](size_t elems, size_t esz) -> void* {
    off = (off + 255) & ~(size_t)255;
    void* p = ws + off;
    off += elems * esz;
    return p;
  };

  const size_t nWih  = (size_t)3 * H_SZ * C_SZ;
  const size_t nWhh  = (size_t)3 * H_SZ * H_SZ;
  const size_t nWout = (size_t)C_SZ * H_SZ;
  const size_t nBH   = (size_t)B_SZ * H_SZ;
  const size_t nBC   = (size_t)B_SZ * C_SZ;

  bf16* wih_hi  = (bf16*)take(nWih, 2);
  bf16* wih_lo  = (bf16*)take(nWih, 2);
  bf16* whh_hi  = (bf16*)take(nWhh, 2);
  bf16* whh_lo  = (bf16*)take(nWhh, 2);
  bf16* wout_hi = (bf16*)take(nWout, 2);
  bf16* wout_lo = (bf16*)take(nWout, 2);
  bf16* nz_hi   = (bf16*)take(nBH, 2);
  bf16* nz_lo   = (bf16*)take(nBH, 2);
  bf16* cod_hi  = (bf16*)take(nBC, 2);
  bf16* cod_lo  = (bf16*)take(nBC, 2);
  float* hf[2]  = { (float*)take(nBH, 4), (float*)take(nBH, 4) };
  bf16* hhi[2]  = { (bf16*)take(nBH, 2), (bf16*)take(nBH, 2) };
  bf16* hlo[2]  = { (bf16*)take(nBH, 2), (bf16*)take(nBH, 2) };

  auto cvt = [&](const float* src, bf16* hi, bf16* lo, size_t n) {
    int blocks = (int)((n + 255) / 256);
    if (blocks > 4096) blocks = 4096;
    cvt_split<<<blocks, 256, 0, stream>>>(src, hi, lo, n);
  };
  cvt(w_ih,  wih_hi,  wih_lo,  nWih);
  cvt(w_hh,  whh_hi,  whh_lo,  nWhh);
  cvt(w_out, wout_hi, wout_lo, nWout);
  cvt(noise, nz_hi,   nz_lo,   nBH);

  zero_h<<<(int)((nBH + 255) / 256), 256, 0, stream>>>(hf[0], hhi[0], hlo[0], (int)nBH);

  dim3 headGrid(C_SZ / 128, B_SZ / 64);   // (32, 4)
  dim3 gruGrid(H_SZ / 32,  B_SZ / 64);    // (16, 4) = 64 blocks
  const size_t sampStride = (size_t)L_SZ * C_SZ;
  const size_t hiddStride = (size_t)L_SZ * H_SZ;

  // samples[:,0,:] = head(noise); codes0 -> ws
  head_kernel<<<headGrid, 256, 0, stream>>>(
      nz_hi, nz_lo, wout_hi, wout_lo, b_out,
      samples, sampStride, cod_hi, cod_lo);

  for (int t = 0; t < L_SZ; ++t) {
    int cur = t & 1, nxt = (t + 1) & 1;
    // h_{t+1} = GRUCell(codes_t, h_t) ; hiddens[:,t,:] = h_{t+1}
    gru_kernel<<<gruGrid, 256, 0, stream>>>(
        cod_hi, cod_lo, wih_hi, wih_lo,
        hhi[cur], hlo[cur], whh_hi, whh_lo,
        b_ih, b_hh, hf[cur],
        hf[nxt], hhi[nxt], hlo[nxt],
        hiddens + (size_t)t * H_SZ, hiddStride);
    if (t + 1 < L_SZ) {
      // samples[:,t+1,:] = codes_{t+1} = head(h_{t+1})
      head_kernel<<<headGrid, 256, 0, stream>>>(
          hhi[nxt], hlo[nxt], wout_hi, wout_lo, b_out,
          samples + (size_t)(t + 1) * C_SZ, sampStride, cod_hi, cod_lo);
    }
  }
}
